// MultiHeadAttention_48395691491998
// MI455X (gfx1250) — matrix operand order repaired and checked by reading, not compiled, not run on hardware
//
#include <hip/hip_runtime.h>
#include <hip/hip_bf16.h>

#define B_  4
#define S_  2048
#define D_  1024
#define H_  16
#define DK_ 64

typedef float  v8f  __attribute__((ext_vector_type(8)));
typedef float  v4f  __attribute__((ext_vector_type(4)));
typedef __bf16 v4bf __attribute__((ext_vector_type(4)));
typedef __bf16 v8bf __attribute__((ext_vector_type(8)));
typedef __bf16 v16bf __attribute__((ext_vector_type(16)));

// ---- fragment loaders (work for global or LDS pointers) ---------------------
// A-matrix 16x32 bf16 (ISA 7.12.2): lane m = l%16; half h = l/16;
//   e[0..7] = A[m][8h + 0..7], e[8..15] = A[m][16 + 8h + 0..7]
__device__ __forceinline__ v16bf load_afrag(const __bf16* row, int half) {
  v8bf lo = *(const v8bf*)(row + half * 8);
  v8bf hi = *(const v8bf*)(row + half * 8 + 16);
  return __builtin_shufflevector(lo, hi, 0,1,2,3,4,5,6,7,8,9,10,11,12,13,14,15);
}
// B-matrix 32x16 bf16: lane n = l%16 holds column n; half h = l/16;
//   e[0..7] = B[8h + 0..7][n], e[8..15] = B[16 + 8h + 0..7][n]. `col` points at B column n contiguous in memory.
__device__ __forceinline__ v16bf load_bfrag(const __bf16* col, int half) {
  v8bf lo = *(const v8bf*)(col + half * 8);
  v8bf hi = *(const v8bf*)(col + half * 8 + 16);
  return __builtin_shufflevector(lo, hi, 0,1,2,3,4,5,6,7,8,9,10,11,12,13,14,15);
}
__device__ __forceinline__ v8f wmma_bf16(v16bf a, v16bf b, v8f c) {
  return __builtin_amdgcn_wmma_f32_16x16x32_bf16(false, a, false, b, (short)0, c,
                                                 false, false);
}
__device__ __forceinline__ unsigned lds_off(const void* p) {
  return (unsigned)(size_t)p;  // low 32 bits of a generic LDS address = ds offset
}
// CDNA5 async global->LDS copy of one 16B chunk (tracked by ASYNCcnt)
__device__ __forceinline__ void async_b128(unsigned ldsoff, const __bf16* g) {
  asm volatile("global_load_async_to_lds_b128 %0, %1, off"
               :: "v"(ldsoff), "v"((unsigned long long)(size_t)g) : "memory");
}

// ---- kernel 1: f32 -> bf16 conversion (vectorized, grid-stride) -------------
__global__ __launch_bounds__(256) void k_cvt(const float* __restrict__ src,
                                             __bf16* __restrict__ dst, int n) {
  int stride = gridDim.x * blockDim.x;
  int m = n >> 2;
  for (int i = blockIdx.x * blockDim.x + threadIdx.x; i < m; i += stride) {
    v4f f = ((const v4f*)src)[i];
    v4bf o;
#pragma unroll
    for (int j = 0; j < 4; ++j) o[j] = (__bf16)f[j];
    ((v4bf*)dst)[i] = o;
  }
}

// ---- kernel 2: QKV projection (x @ W^T), 32x64 register tile per wave -------
// grid = (B*S/32, D/512, 3), block = 256 (8 waves). Each wave's 64 output
// columns are exactly one head's DK -> trivial head-split / V-transpose stores.
__global__ __launch_bounds__(256) void k_proj(const __bf16* __restrict__ xq,
                                              const __bf16* __restrict__ xk,
                                              const __bf16* __restrict__ xv,
                                              const __bf16* __restrict__ wbf,
                                              __bf16* __restrict__ qh,
                                              __bf16* __restrict__ kh,
                                              __bf16* __restrict__ vt) {
  const int wave = threadIdx.x >> 5, lane = threadIdx.x & 31;
  const int half = lane >> 4, ln = lane & 15;
  const int rowTile = blockIdx.x;                 // 32 rows over B*S
  const int col0 = blockIdx.y * 512 + wave * 64;  // 64 cols = one head
  const int proj = blockIdx.z;

  const __bf16* x = (proj == 0) ? xq : (proj == 1) ? xk : xv;
  const __bf16* w = wbf + (size_t)proj * D_ * D_;

  const __bf16* arow0 = x + (size_t)(rowTile * 32 + ln) * D_;
  const __bf16* arow1 = arow0 + (size_t)16 * D_;

  v8f acc[2][4];
#pragma unroll
  for (int rt = 0; rt < 2; ++rt)
#pragma unroll
    for (int c = 0; c < 4; ++c) acc[rt][c] = (v8f){};

  for (int k0 = 0; k0 < D_; k0 += 32) {
    v16bf a0 = load_afrag(arow0 + k0, half);
    v16bf a1 = load_afrag(arow1 + k0, half);
#pragma unroll
    for (int c = 0; c < 4; ++c) {
      v16bf b = load_bfrag(w + (size_t)(col0 + c * 16 + ln) * D_ + k0, half);
      acc[0][c] = wmma_bf16(a0, b, acc[0][c]);
      acc[1][c] = wmma_bf16(a1, b, acc[1][c]);
    }
  }

  const int h = col0 >> 6;
#pragma unroll
  for (int rt = 0; rt < 2; ++rt)
#pragma unroll
    for (int r = 0; r < 8; ++r) {
      int gr = rowTile * 32 + rt * 16 + r + 8 * half;  // row over B*S
      int bb = gr >> 11, ss = gr & (S_ - 1);
#pragma unroll
      for (int c = 0; c < 4; ++c) {
        int dk = c * 16 + ln;
        __bf16 val = (__bf16)acc[rt][c][r];
        if (proj == 2)
          vt[(((size_t)bb * H_ + h) * DK_ + dk) * S_ + ss] = val;
        else if (proj == 0)
          qh[(((size_t)bb * H_ + h) * S_ + ss) * DK_ + dk] = val;
        else
          kh[(((size_t)bb * H_ + h) * S_ + ss) * DK_ + dk] = val;
      }
    }
}

// ---- kernel 3: block-level causal flash attention ---------------------------
// grid = (B*H, S/128), block = 256. 8 waves share one (b,h); wave w owns
// query tile qbase+16w. K/V 32-key tiles are async-staged into double-buffered
// LDS once per block (8x traffic reduction vs per-wave streaming).
__global__ __launch_bounds__(256) void k_attn(const __bf16* __restrict__ qh,
                                              const __bf16* __restrict__ kh,
                                              const __bf16* __restrict__ vt,
                                              __bf16* __restrict__ aout) {
  __shared__ __align__(16) __bf16 kbuf[2][32][64];  // 32 keys x 64 dk
  __shared__ __align__(16) __bf16 vbuf[2][64][32];  // 64 dk x 32 keys (V^T)
  __shared__ __align__(16) __bf16 plds[8][16][32];  // per-wave P tile

  const int tid = threadIdx.x;
  const int wave = tid >> 5, lane = tid & 31;
  const int half = lane >> 4, ln = lane & 15;
  const int bh = blockIdx.x;
  const int qbase = blockIdx.y * 128;
  const int q0 = qbase + wave * 16;

  const __bf16* qb = qh + (size_t)bh * S_ * DK_;
  const __bf16* kb = kh + (size_t)bh * S_ * DK_;
  const __bf16* vb = vt + (size_t)bh * DK_ * S_;

  // Q fragments (A layout), kdim 0..31 and 32..63
  const __bf16* qrow = qb + (size_t)(q0 + ln) * DK_;
  v16bf qa0 = load_afrag(qrow, half);
  v16bf qa1 = load_afrag(qrow + 32, half);

  float mrow[8], lrow[8];
  v8f o[4];
#pragma unroll
  for (int r = 0; r < 8; ++r) { mrow[r] = -1e30f; lrow[r] = 0.f; }
#pragma unroll
  for (int c = 0; c < 4; ++c) o[c] = (v8f){};

  // cooperative async staging: 256 threads x (one 16B K chunk + one 16B V chunk)
  auto stage = [&](int buf, int k0) {
    async_b128(lds_off(&kbuf[buf][tid >> 3][(tid & 7) * 8]),
               kb + (size_t)(k0 + (tid >> 3)) * DK_ + (tid & 7) * 8);
    async_b128(lds_off(&vbuf[buf][tid >> 2][(tid & 3) * 8]),
               vb + (size_t)(tid >> 2) * S_ + k0 + (tid & 3) * 8);
  };

  const int niter = (qbase + 128) >> 5;  // keys 0 .. qbase+127, uniform per block
  stage(0, 0);
  for (int it = 0; it < niter; ++it) {
    const int cur = it & 1;
    const int k0 = it * 32;
    if (it + 1 < niter) {
      stage(1 - cur, k0 + 32);
      asm volatile("s_wait_asynccnt 0x2" ::: "memory");  // cur buffer landed
    } else {
      asm volatile("s_wait_asynccnt 0x0" ::: "memory");
    }
    __syncthreads();  // all threads' chunks visible

    // ---- scores: Q(16x64) @ K^T from LDS ----
    v8f s[2];
#pragma unroll
    for (int t = 0; t < 2; ++t) {
      const __bf16* krow = &kbuf[cur][t * 16 + ln][0];
      v8f sc = {};
      sc = wmma_bf16(qa0, load_bfrag(krow, half), sc);
      sc = wmma_bf16(qa1, load_bfrag(krow + 32, half), sc);
      s[t] = sc;
    }
    // ---- scale + causal mask (over-causal iters become all-masked: inert) ----
#pragma unroll
    for (int t = 0; t < 2; ++t)
#pragma unroll
      for (int r = 0; r < 8; ++r) {
        float v = s[t][r] * 0.125f;  // 1/sqrt(64)
        if (k0 + t * 16 + ln > q0 + r + 8 * half) v = -1e30f;
        s[t][r] = v;
      }
    // ---- row max over 32 keys (16-lane reduction inside half-wave) ----
    float mx[8];
#pragma unroll
    for (int r = 0; r < 8; ++r) mx[r] = fmaxf(s[0][r], s[1][r]);
#pragma unroll
    for (int off = 8; off; off >>= 1)
#pragma unroll
      for (int r = 0; r < 8; ++r) mx[r] = fmaxf(mx[r], __shfl_xor(mx[r], off, 32));
    // ---- online softmax update ----
    float alpha[8], rs[8];
#pragma unroll
    for (int r = 0; r < 8; ++r) {
      float mn = fmaxf(mrow[r], mx[r]);
      alpha[r] = __expf(mrow[r] - mn);
      mrow[r] = mn;
      float p0 = __expf(s[0][r] - mn);
      float p1 = __expf(s[1][r] - mn);
      s[0][r] = p0; s[1][r] = p1;
      rs[r] = p0 + p1;
    }
#pragma unroll
    for (int off = 8; off; off >>= 1)
#pragma unroll
      for (int r = 0; r < 8; ++r) rs[r] += __shfl_xor(rs[r], off, 32);
#pragma unroll
    for (int r = 0; r < 8; ++r) lrow[r] = lrow[r] * alpha[r] + rs[r];
#pragma unroll
    for (int c = 0; c < 4; ++c)
#pragma unroll
      for (int r = 0; r < 8; ++r) o[c][r] *= alpha[r];

    // ---- C-layout P -> LDS -> A-fragment (same-wave DS ops are in-order) ----
#pragma unroll
    for (int t = 0; t < 2; ++t)
#pragma unroll
      for (int r = 0; r < 8; ++r)
        plds[wave][r + 8 * half][t * 16 + ln] = (__bf16)s[t][r];
    v16bf pa = load_afrag(&plds[wave][ln][0], half);

    // ---- O += P(16x32) @ V(32x64), V^T tiles from LDS ----
#pragma unroll
    for (int c = 0; c < 4; ++c)
      o[c] = wmma_bf16(pa, load_bfrag(&vbuf[cur][c * 16 + ln][0], half), o[c]);

    __syncthreads();  // done reading cur before it is overwritten
  }

  // ---- epilogue: normalize, store [B,S,D] bf16 ----
  const int b = bh >> 4, h = bh & (H_ - 1);
#pragma unroll
  for (int r = 0; r < 8; ++r) {
    float inv = 1.0f / lrow[r];
    int srow = q0 + r + 8 * half;
#pragma unroll
    for (int c = 0; c < 4; ++c)
      aout[((size_t)(b * S_ + srow)) * D_ + h * 64 + c * 16 + ln] =
          (__bf16)(o[c][r] * inv);
  }
}

// ---- kernel 4: output projection attn_out @ Wo^T -> f32, 32x64 tiles --------
// grid = (B*S/32, D/512), block = 256.
__global__ __launch_bounds__(256) void k_oproj(const __bf16* __restrict__ ain,
                                               const __bf16* __restrict__ wo,
                                               float* __restrict__ out) {
  const int wave = threadIdx.x >> 5, lane = threadIdx.x & 31;
  const int half = lane >> 4, ln = lane & 15;
  const int rowTile = blockIdx.x;
  const int col0 = blockIdx.y * 512 + wave * 64;

  const __bf16* arow0 = ain + (size_t)(rowTile * 32 + ln) * D_;
  const __bf16* arow1 = arow0 + (size_t)16 * D_;

  v8f acc[2][4];
#pragma unroll
  for (int rt = 0; rt < 2; ++rt)
#pragma unroll
    for (int c = 0; c < 4; ++c) acc[rt][c] = (v8f){};

  for (int k0 = 0; k0 < D_; k0 += 32) {
    v16bf a0 = load_afrag(arow0 + k0, half);
    v16bf a1 = load_afrag(arow1 + k0, half);
#pragma unroll
    for (int c = 0; c < 4; ++c) {
      v16bf b = load_bfrag(wo + (size_t)(col0 + c * 16 + ln) * D_ + k0, half);
      acc[0][c] = wmma_bf16(a0, b, acc[0][c]);
      acc[1][c] = wmma_bf16(a1, b, acc[1][c]);
    }
  }
#pragma unroll
  for (int rt = 0; rt < 2; ++rt)
#pragma unroll
    for (int r = 0; r < 8; ++r) {
      size_t gr = rowTile * 32 + rt * 16 + r + 8 * half;
#pragma unroll
      for (int c = 0; c < 4; ++c)
        out[gr * D_ + col0 + c * 16 + ln] = acc[rt][c][r];
    }
}

// ---- host entry -------------------------------------------------------------
extern "C" void kernel_launch(void* const* d_in, const int* in_sizes, int n_in,
                              void* d_out, int out_size, void* d_ws, size_t ws_size,
                              hipStream_t stream) {
  const float* q  = (const float*)d_in[0];
  const float* k  = (const float*)d_in[1];
  const float* v  = (const float*)d_in[2];
  const float* Wq = (const float*)d_in[3];
  const float* Wk = (const float*)d_in[4];
  const float* Wv = (const float*)d_in[5];
  const float* Wo = (const float*)d_in[6];
  float* out = (float*)d_out;

  char* ws = (char*)d_ws;
  const size_t nW  = (size_t)D_ * D_;              // one weight matrix
  const size_t nX  = (size_t)B_ * S_ * D_;         // one activation tensor
  __bf16* wbf = (__bf16*)ws;  ws += 4 * nW * 2;    // Wq,Wk,Wv,Wo bf16
  __bf16* xqb = (__bf16*)ws;  ws += nX * 2;
  __bf16* xkb = (__bf16*)ws;  ws += nX * 2;
  __bf16* xvb = (__bf16*)ws;  ws += nX * 2;
  __bf16* qhb = (__bf16*)ws;  ws += nX * 2;        // [B,H,S,DK]
  __bf16* khb = (__bf16*)ws;  ws += nX * 2;        // [B,H,S,DK]
  __bf16* vtb = (__bf16*)ws;  ws += nX * 2;        // [B,H,DK,S]
  __bf16* ahb = (__bf16*)ws;                       // [B,S,D]

  k_cvt<<<1024, 256, 0, stream>>>(q, xqb, (int)nX);
  k_cvt<<<1024, 256, 0, stream>>>(k, xkb, (int)nX);
  k_cvt<<<1024, 256, 0, stream>>>(v, xvb, (int)nX);
  k_cvt<<<256, 256, 0, stream>>>(Wq, wbf + 0 * nW, (int)nW);
  k_cvt<<<256, 256, 0, stream>>>(Wk, wbf + 1 * nW, (int)nW);
  k_cvt<<<256, 256, 0, stream>>>(Wv, wbf + 2 * nW, (int)nW);
  k_cvt<<<256, 256, 0, stream>>>(Wo, wbf + 3 * nW, (int)nW);

  k_proj<<<dim3(B_ * S_ / 32, D_ / 512, 3), 256, 0, stream>>>(
      xqb, xkb, xvb, wbf, qhb, khb, vtb);
  k_attn<<<dim3(B_ * H_, S_ / 128), 256, 0, stream>>>(qhb, khb, vtb, ahb);
  k_oproj<<<dim3(B_ * S_ / 32, D_ / 512), 256, 0, stream>>>(
      ahb, wbf + 3 * nW, out);
}
